// ManualRNN_76166950028403
// MI455X (gfx1250) — matrix operand order repaired and checked by reading, not compiled, not run on hardware
//
#include <hip/hip_runtime.h>
#include <hip/hip_bf16.h>
#include <math.h>

// ---------------------------------------------------------------------------
// ManualRNN on MI455X (gfx1250): bf16 WMMA GEMMs, f32 accumulation.
// Staging: A tile via per-lane GLOBAL_LOAD_ASYNC_TO_LDS_B128 (ASYNCcnt),
//          B tile via one TENSOR_LOAD_TO_LDS DMA per workgroup (TENSORcnt),
//          double-buffered LDS, waits pipelined one chunk deep.
//   pre:     x -> bf16
//   phase 0: xproj = x @ Wxh + bh              (M=65536, N=1024, K=512)
//   phase 1: 256 x  h = tanh(xproj_t + h@Whh)  (M=256,  N=1024, K=1024)
//   phase 2: out = h @ fc_w.T + fc_b           (M=256,  N=1000, K=1024)
// ---------------------------------------------------------------------------

typedef __bf16 bf16_t;
typedef __attribute__((ext_vector_type(16))) __bf16 v16bf;
typedef __attribute__((ext_vector_type(8)))  __bf16 v8bf;
typedef __attribute__((ext_vector_type(4)))  __bf16 v4bf;
typedef __attribute__((ext_vector_type(8)))  float  v8f;
typedef __attribute__((ext_vector_type(4)))  unsigned int v4u;
typedef __attribute__((ext_vector_type(8)))  int v8i;
typedef __attribute__((ext_vector_type(4)))  int v4i;

#define RNN_B  256
#define RNN_S  256
#define RNN_I  512
#define RNN_H  1024
#define RNN_C  1000
#define RNN_CP 1024   // classes padded to 1024 (rows 1000..1023 zero-filled)

#if defined(__has_builtin)
#  if __has_builtin(__builtin_amdgcn_tensor_load_to_lds)
#    define RNN_HAS_TDM 1
#  else
#    define RNN_HAS_TDM 0
#  endif
#else
#  define RNN_HAS_TDM 0
#endif

union V16u { v16bf v; v8bf h[2]; };

// ---- CDNA5 async copy: global -> LDS, 16 bytes per lane, ASYNCcnt-tracked.
__device__ __forceinline__ void async_copy_b128(bf16_t* lds_dst, const bf16_t* gsrc) {
  // low 32 bits of a generic LDS pointer = workgroup-relative LDS byte address
  uint32_t laddr = (uint32_t)(uintptr_t)lds_dst;
  asm volatile("global_load_async_to_lds_b128 %0, %1, off"
               :: "v"(laddr), "v"(gsrc)
               : "memory");
}

template <int N>
__device__ __forceinline__ void wait_asynccnt() {
#if __has_builtin(__builtin_amdgcn_s_wait_asynccnt)
  __builtin_amdgcn_s_wait_asynccnt(N);
#else
  asm volatile("s_wait_asynccnt %0" :: "i"(N) : "memory");
#endif
}

template <int N>
__device__ __forceinline__ void wait_tensorcnt() {
#if __has_builtin(__builtin_amdgcn_s_wait_tensorcnt)
  __builtin_amdgcn_s_wait_tensorcnt(N);
#else
  asm volatile("s_wait_tensorcnt %0" :: "i"(N) : "memory");
#endif
}

#if RNN_HAS_TDM
// ---- TDM: DMA one [256 rows x 32 elems] bf16 tile (row stride K elems) from
// global into contiguous LDS. One instruction per wave; TENSORcnt-tracked.
// Descriptor bit layout per CDNA5 ISA ch.8 (D# groups 0/1; 2D tile -> rest 0).
// clang-23 toolchain form: 6 args (g0 u32x4, g1 i32x8, i32x4, i32x4, i32x8, cpol).
__device__ __forceinline__ void tdm_load_b_tile(const bf16_t* gsrc,
                                                bf16_t* lds_dst, int K) {
  uint64_t ga = (uint64_t)(uintptr_t)gsrc;
  uint32_t la = (uint32_t)(uintptr_t)lds_dst;
  v4u g0;
  g0[0] = 1u;                                            // count=1 (valid user D#)
  g0[1] = la;                                            // lds_addr [63:32]
  g0[2] = (uint32_t)ga;                                  // global_addr [95:64]
  g0[3] = (uint32_t)((ga >> 32) & 0x1FFFFFFu) | (2u << 30); // addr[56:32] | type=2
  v8i g1;
  g1[0] = 0x00010000;                                    // data_size=2B, no flags
  g1[1] = (K & 0xFFFF) << 16;                            // tensor_dim0[15:0]
  g1[2] = ((K >> 16) & 0xFFFF) | ((RNN_H & 0xFFFF) << 16); // dim0 hi | tensor_dim1 lo
  g1[3] = 32 << 16;                                      // dim1 hi=0 | tile_dim0=32
  g1[4] = 256;                                           // tile_dim1=256, tile_dim2=0
  g1[5] = K;                                             // tensor_dim0_stride[31:0]
  g1[6] = 0;                                             // stride0 hi | dim1_stride lo
  g1[7] = 0;
  v4i z4; z4[0] = 0; z4[1] = 0; z4[2] = 0; z4[3] = 0;
  v8i z8; z8[0] = 0; z8[1] = 0; z8[2] = 0; z8[3] = 0;
  z8[4] = 0; z8[5] = 0; z8[6] = 0; z8[7] = 0;
  __builtin_amdgcn_tensor_load_to_lds(g0, g1, z4, z4, z8, 0);
}
#endif

// A fragment (16x32 bf16): lane l (0-15) holds row l; lo-half lanes carry
// K 0..15, hi-half lanes K 16..31 (matches B fragment order).
__device__ __forceinline__ v16bf load_a_frag(const bf16_t* As, int row, int hi) {
  const bf16_t* r = As + row * 32;
  V16u u;
  u.h[0] = *(const v8bf*)(r + hi * 16);
  u.h[1] = *(const v8bf*)(r + hi * 16 + 8);
  return u.v;
}

// B fragment (32x16 bf16): lane l holds column l; lo-half lanes K 0..15,
// hi-half lanes K 16..31. Bs is N-major [rows][32] -> contiguous 32B read.
__device__ __forceinline__ v16bf load_b_frag(const bf16_t* Bs, int col, int hi) {
  const bf16_t* r = Bs + col * 32 + hi * 16;
  V16u u;
  u.h[0] = *(const v8bf*)(r);
  u.h[1] = *(const v8bf*)(r + 8);
  return u.v;
}

__device__ __forceinline__ v8f wmma_bf16(v16bf a, v16bf b, v8f c) {
  return __builtin_amdgcn_wmma_f32_16x16x32_bf16(
      /*neg_a=*/false, a, /*neg_b=*/false, b,
      /*c_mod=*/(short)0, c, /*reuse_a=*/false, /*reuse_b=*/false);
}

// MODE 0: +bias(bh), store bf16 xproj
// MODE 1: + xproj[:,t,:], tanh, store bf16 h_next
// MODE 2: +fc_b, store f32 out with N<1000 guard
// 256 threads = 8 waves in 2(M) x 4(N); tile 64(M) x 256(N);
// each wave: 32x64 strip = 8 WMMA accumulators; K staged in 32-chunks.
template <int MODE>
__global__ __launch_bounds__(256) void rnn_gemm_wmma(
    const bf16_t* __restrict__ Ap, const bf16_t* __restrict__ Bp,
    const float* __restrict__ bias, const bf16_t* __restrict__ addend,
    int t, void* __restrict__ outp, int K)
{
  __shared__ __attribute__((aligned(16))) bf16_t As[2][64 * 32];    // 2 x 4 KB
  __shared__ __attribute__((aligned(16))) bf16_t Bs[2][256 * 32];   // 2 x 16 KB

  const int tid  = threadIdx.x;
  const int n0   = blockIdx.x * 256;
  const int m0   = blockIdx.y * 64;
  const int lane = tid & 31;
  const int wid  = tid >> 5;
  const int l    = lane & 15;
  const int hi   = lane >> 4;
  const int wm   = (wid & 1) * 32;    // wave row offset in 64-row tile
  const int wn   = (wid >> 1) * 64;   // wave col offset in 256-col tile

  // A staging: 64x32 tile, one b128 async copy per thread
  const int aRow = tid >> 2, aOff = (tid & 3) * 8;
  const bf16_t* aSrc  = Ap + (size_t)(m0 + aRow) * K + aOff;
  const bf16_t* bTile = Bp + (size_t)n0 * K;       // B tile origin (row n0, k=0)
#if !RNN_HAS_TDM
  const bf16_t* bSrc  = Bp + (size_t)(n0 + tid) * K;
#endif

  v8f acc[8] = {};   // acc[mi*4+j]: rows wm+mi*16.., cols wn+j*16..

  const int nChunks = K / 32;

  // prime the pipeline: stage chunk 0 into buffer 0
  async_copy_b128(&As[0][aRow * 32 + aOff], aSrc);
#if RNN_HAS_TDM
  if (wid == 0) tdm_load_b_tile(bTile, &Bs[0][0], K);
#else
#pragma unroll
  for (int q = 0; q < 4; ++q)
    async_copy_b128(&Bs[0][tid * 32 + q * 8], bSrc + q * 8);
#endif

  for (int c = 0; c < nChunks; ++c) {
    const int cur = c & 1;
    if (c + 1 < nChunks) {
      const int k1 = (c + 1) * 32;
      async_copy_b128(&As[cur ^ 1][aRow * 32 + aOff], aSrc + k1);
#if RNN_HAS_TDM
      if (wid == 0) tdm_load_b_tile(bTile + k1, &Bs[cur ^ 1][0], K);
      wait_asynccnt<1>();   // A(c) done; A(c+1) in flight
      wait_tensorcnt<1>();  // B(c) done; B(c+1) in flight (no-op for waves 1..7)
#else
#pragma unroll
      for (int q = 0; q < 4; ++q)
        async_copy_b128(&Bs[cur ^ 1][tid * 32 + q * 8], bSrc + k1 + q * 8);
      wait_asynccnt<5>();
#endif
    } else {
      wait_asynccnt<0>();
#if RNN_HAS_TDM
      wait_tensorcnt<0>();
#endif
    }
    __syncthreads();        // all waves' DMA'd LDS data visible

    // hoist all fragment loads so ds_loads clause and WMMAs issue back-to-back
    v16bf a0 = load_a_frag(&As[cur][0], wm + l, hi);
    v16bf a1 = load_a_frag(&As[cur][0], wm + 16 + l, hi);
    v16bf b0 = load_b_frag(&Bs[cur][0], wn + 0 * 16 + l, hi);
    v16bf b1 = load_b_frag(&Bs[cur][0], wn + 1 * 16 + l, hi);
    v16bf b2 = load_b_frag(&Bs[cur][0], wn + 2 * 16 + l, hi);
    v16bf b3 = load_b_frag(&Bs[cur][0], wn + 3 * 16 + l, hi);

    acc[0] = wmma_bf16(a0, b0, acc[0]);
    acc[1] = wmma_bf16(a0, b1, acc[1]);
    acc[2] = wmma_bf16(a0, b2, acc[2]);
    acc[3] = wmma_bf16(a0, b3, acc[3]);
    acc[4] = wmma_bf16(a1, b0, acc[4]);
    acc[5] = wmma_bf16(a1, b1, acc[5]);
    acc[6] = wmma_bf16(a1, b2, acc[6]);
    acc[7] = wmma_bf16(a1, b3, acc[7]);

    __syncthreads();        // protect buffer reuse before next chunk's DMA
  }

  // epilogue: C/D layout — VGPR v holds row (v + 8*hi), column l.
#pragma unroll
  for (int mi = 0; mi < 2; ++mi) {
#pragma unroll
    for (int j = 0; j < 4; ++j) {
#pragma unroll
      for (int v = 0; v < 8; ++v) {
        const int gm = m0 + wm + mi * 16 + hi * 8 + v;
        const int gn = n0 + wn + j * 16 + l;
        float val = acc[mi * 4 + j][v];
        if (MODE == 0) {
          val += bias[gn];
          ((bf16_t*)outp)[(size_t)gm * RNN_H + gn] = (bf16_t)val;
        } else if (MODE == 1) {
          val += (float)addend[((size_t)gm * RNN_S + t) * RNN_H + gn];
          val = tanhf(val);
          ((bf16_t*)outp)[(size_t)gm * RNN_H + gn] = (bf16_t)val;
        } else {
          if (gn < RNN_C)
            ((float*)outp)[(size_t)gm * RNN_C + gn] = val + bias[gn];
        }
      }
    }
  }
}

// out[c][r] = bf16(in[r][c]) : [R][C] f32 -> [C][R] bf16 (N-major for WMMA B)
__global__ __launch_bounds__(256) void rnn_transpose_to_bf16(
    const float* __restrict__ in, bf16_t* __restrict__ out, int R, int C)
{
  const int idx = blockIdx.x * 256 + threadIdx.x;
  const int r = idx / C, c = idx % C;
  out[(size_t)c * R + r] = (bf16_t)in[idx];
}

// fc_w is already [classes][K] = N-major; convert and zero-pad rows >= 1000
__global__ __launch_bounds__(256) void rnn_fcw_to_bf16(
    const float* __restrict__ in, bf16_t* __restrict__ out)
{
  const int idx = blockIdx.x * 256 + threadIdx.x;
  const int c = idx >> 10;  // / RNN_H
  out[idx] = (c < RNN_C) ? (bf16_t)in[idx] : (bf16_t)0.0f;
}

// x (f32, contiguous) -> bf16, 4 elements/thread
__global__ __launch_bounds__(256) void rnn_f32_to_bf16(
    const float* __restrict__ in, bf16_t* __restrict__ out)
{
  const size_t i = ((size_t)blockIdx.x * 256 + threadIdx.x) * 4;
  float4 f = *(const float4*)(in + i);
  v4bf c;
  c[0] = (bf16_t)f.x; c[1] = (bf16_t)f.y; c[2] = (bf16_t)f.z; c[3] = (bf16_t)f.w;
  *(v4bf*)(out + i) = c;
}

__global__ __launch_bounds__(256) void rnn_zero_bf16(bf16_t* __restrict__ p)
{
  p[(size_t)blockIdx.x * 256 + threadIdx.x] = (bf16_t)0.0f;
}

extern "C" void kernel_launch(void* const* d_in, const int* in_sizes, int n_in,
                              void* d_out, int out_size, void* d_ws, size_t ws_size,
                              hipStream_t stream)
{
  (void)in_sizes; (void)n_in; (void)out_size; (void)ws_size;
  const float* x    = (const float*)d_in[0];   // [B,S,I]
  const float* Wxh  = (const float*)d_in[1];   // [I,H]
  const float* Whh  = (const float*)d_in[2];   // [H,H]
  const float* bh   = (const float*)d_in[3];   // [H]
  const float* fc_w = (const float*)d_in[4];   // [C,H]
  const float* fc_b = (const float*)d_in[5];   // [C]

  // Workspace layout (bf16): ~205 MB total
  char* ws = (char*)d_ws;
  size_t off = 0;
  bf16_t* xbf   = (bf16_t*)(ws + off); off += (size_t)RNN_B * RNN_S * RNN_I * 2;  //  64 MB
  bf16_t* xproj = (bf16_t*)(ws + off); off += (size_t)RNN_B * RNN_S * RNN_H * 2;  // 134 MB
  bf16_t* WxhT  = (bf16_t*)(ws + off); off += (size_t)RNN_H * RNN_I * 2;          //   1 MB
  bf16_t* WhhT  = (bf16_t*)(ws + off); off += (size_t)RNN_H * RNN_H * 2;          //   2 MB
  bf16_t* fcw   = (bf16_t*)(ws + off); off += (size_t)RNN_CP * RNN_H * 2;         //   2 MB
  bf16_t* h0    = (bf16_t*)(ws + off); off += (size_t)RNN_B * RNN_H * 2;
  bf16_t* h1    = (bf16_t*)(ws + off); off += (size_t)RNN_B * RNN_H * 2;

  // conversions (streaming, negligible vs GEMMs)
  rnn_f32_to_bf16<<<(RNN_B * RNN_S * RNN_I) / (256 * 4), 256, 0, stream>>>(x, xbf);
  rnn_transpose_to_bf16<<<(RNN_I * RNN_H) / 256, 256, 0, stream>>>(Wxh, WxhT, RNN_I, RNN_H);
  rnn_transpose_to_bf16<<<(RNN_H * RNN_H) / 256, 256, 0, stream>>>(Whh, WhhT, RNN_H, RNN_H);
  rnn_fcw_to_bf16<<<(RNN_CP * RNN_H) / 256, 256, 0, stream>>>(fc_w, fcw);
  rnn_zero_bf16<<<(RNN_B * RNN_H) / 256, 256, 0, stream>>>(h0);

  // Phase 0: xproj = x @ Wxh + bh   (grid 4 x 1024)
  rnn_gemm_wmma<0><<<dim3(RNN_H / 256, (RNN_B * RNN_S) / 64), 256, 0, stream>>>(
      xbf, WxhT, bh, nullptr, 0, xproj, RNN_I);

  // Phase 1: sequential scan, one launch per timestep (stream = grid barrier)
  bf16_t* hc = h0; bf16_t* hn = h1;
  for (int t = 0; t < RNN_S; ++t) {
    rnn_gemm_wmma<1><<<dim3(RNN_H / 256, RNN_B / 64), 256, 0, stream>>>(
        hc, WhhT, nullptr, xproj, t, hn, RNN_H);
    bf16_t* tmp = hc; hc = hn; hn = tmp;
  }

  // Phase 2: logits = h @ fc_w.T + fc_b  (N padded to 1024, store-guarded)
  rnn_gemm_wmma<2><<<dim3(RNN_CP / 256, RNN_B / 64), 256, 0, stream>>>(
      hc, fcw, fc_b, nullptr, 0, d_out, RNN_H);
}
